// GNNPowerFlow_60653528154493
// MI455X (gfx1250) — compile-verified, matrix operand order checked by reading, not run on hardware
//
#include <hip/hip_runtime.h>
#include <hip/hip_bf16.h>

// ---------------- problem constants (match reference) ----------------
#define N_BUS   1024
#define BATCH   4096
#define N_NODES (N_BUS * BATCH)      // 4,194,304
#define N_EDGES (4 * N_NODES)        // 16,777,216
#define F_IN    7
#define F1      12
#define F2      12
#define HID     128
#define K1      (N_BUS * F2)         // 12288, K of GEMM1
#define NOUT    (N_BUS * 2)          // 2048,  N of GEMM2
#define BN_EPS  1e-5f

typedef __attribute__((ext_vector_type(16))) _Float16 v16h;
typedef __attribute__((ext_vector_type(8)))  _Float16 v8h;
typedef __attribute__((ext_vector_type(8)))  float    v8f;

// ---------------- workspace layout (bytes, 256-B aligned) ----------------
// [0)                agg (N_NODES*12 f32)  -- reused later as h2 f16 (needs half)
// [OFF_H1)           h1  (N_NODES*12 f32)  -- conv1 out, overwritten by conv2 out
// [OFF_W1H)          W1 as f16 (HID*K1)
// [OFF_W2H)          W2 as f16 (NOUT*HID)
// [OFF_MLP1)         MLP hidden as f16 (BATCH*HID)
// [OFF_STATS)        BN stats: sum1[12] ssq1[12] sum2[12] ssq2[12]
#define OFF_AGG   ((size_t)0)
#define OFF_H1    ((size_t)N_NODES * 12 * 4)                       // 201326592
#define OFF_W1H   (OFF_H1 + (size_t)N_NODES * 12 * 4)              // 402653184
#define OFF_W2H   (OFF_W1H + (size_t)HID * K1 * 2)                 // +3145728
#define OFF_MLP1  (OFF_W2H + (size_t)NOUT * HID * 2)               // +524288
#define OFF_STATS (OFF_MLP1 + (size_t)BATCH * HID * 2)             // +1048576

// ---------------- utility kernels ----------------
__global__ void zero_f32_kernel(float* __restrict__ p, size_t n) {
  size_t i = (size_t)blockIdx.x * blockDim.x + threadIdx.x;
  size_t stride = (size_t)gridDim.x * blockDim.x;
  for (; i < n; i += stride) p[i] = 0.0f;
}

__global__ void cvt_f32_to_f16_kernel(const float* __restrict__ in,
                                      _Float16* __restrict__ out, size_t n) {
  size_t i = (size_t)blockIdx.x * blockDim.x + threadIdx.x;
  size_t stride = (size_t)gridDim.x * blockDim.x;
  for (; i < n; i += stride) out[i] = (_Float16)in[i];
}

// ---------------- edge scatter: agg[dst] += feat[src] ----------------
template <int F>
__global__ void scatter_add_kernel(const float* __restrict__ feat,
                                   const long long* __restrict__ src,
                                   const long long* __restrict__ dst,
                                   float* __restrict__ agg) {
  int e = blockIdx.x * blockDim.x + threadIdx.x;
  if (e >= N_EDGES) return;
  int s = (int)src[e];
  int d = (int)dst[e];
  const float* fs = feat + (size_t)s * F;
  float* ad = agg + (size_t)d * F;
#pragma unroll
  for (int f = 0; f < F; ++f) atomicAdd(ad + f, fs[f]);
}

// ------- conv linear: out = Wrel@agg + brel + Wroot@x ; accumulate BN stats -------
template <int FIN, int FOUT>
__global__ void conv_linear_kernel(const float* __restrict__ agg,
                                   const float* __restrict__ xin,
                                   const float* __restrict__ Wrel,
                                   const float* __restrict__ brel,
                                   const float* __restrict__ Wroot,
                                   float* __restrict__ out,          // may alias xin (row-wise safe)
                                   float* __restrict__ stats) {      // sum[FOUT], ssq[FOUT]
  __shared__ float s_wrel[FOUT * FIN];
  __shared__ float s_wroot[FOUT * FIN];
  __shared__ float s_brel[FOUT];
  __shared__ float s_red[2 * FOUT];
  for (int i = threadIdx.x; i < FOUT * FIN; i += blockDim.x) {
    s_wrel[i]  = Wrel[i];
    s_wroot[i] = Wroot[i];
  }
  if (threadIdx.x < FOUT) s_brel[threadIdx.x] = brel[threadIdx.x];
  if (threadIdx.x < 2 * FOUT) s_red[threadIdx.x] = 0.0f;
  __syncthreads();

  float lsum[FOUT];
  float lssq[FOUT];
#pragma unroll
  for (int q = 0; q < FOUT; ++q) { lsum[q] = 0.0f; lssq[q] = 0.0f; }

  size_t stride = (size_t)gridDim.x * blockDim.x;
  for (size_t i = (size_t)blockIdx.x * blockDim.x + threadIdx.x; i < (size_t)N_NODES; i += stride) {
    float a[FIN], xr[FIN];
    const float* ap = agg + i * FIN;
    const float* xp = xin + i * FIN;
#pragma unroll
    for (int f = 0; f < FIN; ++f) { a[f] = ap[f]; xr[f] = xp[f]; }
    float o[FOUT];
#pragma unroll
    for (int q = 0; q < FOUT; ++q) {
      float acc = s_brel[q];
#pragma unroll
      for (int f = 0; f < FIN; ++f)
        acc = fmaf(s_wrel[q * FIN + f], a[f], fmaf(s_wroot[q * FIN + f], xr[f], acc));
      o[q] = acc;
      lsum[q] += acc;
      lssq[q] += acc * acc;
    }
    float* op = out + i * FOUT;
#pragma unroll
    for (int q = 0; q < FOUT; ++q) op[q] = o[q];
  }

  __syncthreads();
#pragma unroll
  for (int q = 0; q < FOUT; ++q) {
    atomicAdd(&s_red[q], lsum[q]);
    atomicAdd(&s_red[FOUT + q], lssq[q]);
  }
  __syncthreads();
  if (threadIdx.x < 2 * FOUT) atomicAdd(&stats[threadIdx.x], s_red[threadIdx.x]);
}

// ------- batchnorm + relu; fp32 in-place or f16 out -------
template <bool TO_F16>
__global__ void bn_relu_kernel(const float* __restrict__ h,
                               const float* __restrict__ stats,  // sum[12], ssq[12]
                               const float* __restrict__ g,
                               const float* __restrict__ b,
                               float* __restrict__ out32,
                               _Float16* __restrict__ out16) {
  __shared__ float s_scale[F1];
  __shared__ float s_shift[F1];
  if (threadIdx.x < F1) {
    const float invN = 1.0f / (float)N_NODES;
    float mu  = stats[threadIdx.x] * invN;
    float var = stats[F1 + threadIdx.x] * invN - mu * mu;
    float sc  = g[threadIdx.x] * rsqrtf(var + BN_EPS);
    s_scale[threadIdx.x] = sc;
    s_shift[threadIdx.x] = b[threadIdx.x] - mu * sc;
  }
  __syncthreads();
  size_t total = (size_t)N_NODES * F1;
  size_t stride = (size_t)gridDim.x * blockDim.x;
  for (size_t i = (size_t)blockIdx.x * blockDim.x + threadIdx.x; i < total; i += stride) {
    int f = (int)(i % F1);
    float v = fmaf(h[i], s_scale[f], s_shift[f]);
    v = v > 0.0f ? v : 0.0f;
    if (TO_F16) out16[i] = (_Float16)v;
    else        out32[i] = v;
  }
}

// ---------------- WMMA GEMM: C[16x16] = A(16xK) x B(KxN)^T tile -----------------
// A: MxK row-major f16 (K contiguous), B: NxK row-major f16 (K contiguous) -> D = A @ B^T.
// Fragment layout per CDNA5 ISA 16-bit A/B tables: lane[3:0] = row, lane[4] = K-half,
// halves 0..7 hold K = kh*8 + 0..7, halves 8..15 hold K = 16 + kh*8 + 0..7
// -> each fragment is two contiguous 16-byte loads.
__device__ inline v8f wmma_row_tile(const _Float16* __restrict__ A,
                                    const _Float16* __restrict__ B,
                                    int K, int mt, int nt, int lane) {
  const int row = lane & 15;
  const int kh  = (lane >> 4) & 1;
  const _Float16* arow = A + (size_t)(mt * 16 + row) * K + kh * 8;
  const _Float16* brow = B + (size_t)(nt * 16 + row) * K + kh * 8;
  v8f c = {0.f, 0.f, 0.f, 0.f, 0.f, 0.f, 0.f, 0.f};
  for (int k0 = 0; k0 < K; k0 += 32) {
    if (k0 + 32 < K) {
      __builtin_prefetch(arow + k0 + 128, 0, 1);
      __builtin_prefetch(brow + k0 + 128, 0, 1);
    }
    union { v16h v; v8h h[2]; } a, b;
    a.h[0] = *(const v8h*)(arow + k0);
    a.h[1] = *(const v8h*)(arow + k0 + 16);
    b.h[0] = *(const v8h*)(brow + k0);
    b.h[1] = *(const v8h*)(brow + k0 + 16);
    c = __builtin_amdgcn_wmma_f32_16x16x32_f16(
        /*neg_a=*/false, a.v, /*neg_b=*/false, b.v,
        /*c_mod=*/(short)0, c, /*reuse_a=*/false, /*reuse_b=*/false);
  }
  return c;
}

// GEMM1: (BATCH x K1) @ (HID x K1)^T + bias, ReLU, store f16. One wave per C tile.
__global__ void gemm1_wmma_kernel(const _Float16* __restrict__ A,
                                  const _Float16* __restrict__ B,
                                  const float* __restrict__ bias,
                                  _Float16* __restrict__ out) {
  const int lane = threadIdx.x & 31;
  const int wave = threadIdx.x >> 5;
  const int w = blockIdx.x * (blockDim.x >> 5) + wave;
  const int NT = HID / 16;                   // 8
  const int mt = w / NT;
  const int nt = w % NT;
  if (mt >= BATCH / 16) return;

  v8f c = wmma_row_tile(A, B, K1, mt, nt, lane);

  // C layout: col n = nt*16 + lane[3:0]; VGPR r -> row m = mt*16 + lane[4]*8 + r.
  const int n  = nt * 16 + (lane & 15);
  const int mh = (lane >> 4) & 1;
  const float bn = bias[n];
#pragma unroll
  for (int r = 0; r < 8; ++r) {
    int m = mt * 16 + mh * 8 + r;
    float v = c[r] + bn;
    v = v > 0.0f ? v : 0.0f;
    out[(size_t)m * HID + n] = (_Float16)v;
  }
}

// GEMM2: (BATCH x HID) @ (NOUT x HID)^T + bias, store f32 to d_out.
__global__ void gemm2_wmma_kernel(const _Float16* __restrict__ A,
                                  const _Float16* __restrict__ B,
                                  const float* __restrict__ bias,
                                  float* __restrict__ out) {
  const int lane = threadIdx.x & 31;
  const int wave = threadIdx.x >> 5;
  const int w = blockIdx.x * (blockDim.x >> 5) + wave;
  const int NT = NOUT / 16;                  // 128
  const int mt = w / NT;
  const int nt = w % NT;
  if (mt >= BATCH / 16) return;

  v8f c = wmma_row_tile(A, B, HID, mt, nt, lane);

  const int n  = nt * 16 + (lane & 15);
  const int mh = (lane >> 4) & 1;
  const float bn = bias[n];
#pragma unroll
  for (int r = 0; r < 8; ++r) {
    int m = mt * 16 + mh * 8 + r;
    out[(size_t)m * NOUT + n] = c[r] + bn;
  }
}

// ---------------- host-side orchestration ----------------
extern "C" void kernel_launch(void* const* d_in, const int* in_sizes, int n_in,
                              void* d_out, int out_size, void* d_ws, size_t ws_size,
                              hipStream_t stream) {
  // Inputs in setup_inputs() order:
  const float*     x      = (const float*)d_in[0];
  const long long* eidx   = (const long long*)d_in[1];   // (2, N_EDGES) int64
  // d_in[2] = batch_size (scalar, compile-time constant here)
  const float* Wrel1  = (const float*)d_in[3];
  const float* brel1  = (const float*)d_in[4];
  const float* Wroot1 = (const float*)d_in[5];
  const float* Wrel2  = (const float*)d_in[6];
  const float* brel2  = (const float*)d_in[7];
  const float* Wroot2 = (const float*)d_in[8];
  const float* g1 = (const float*)d_in[9];
  const float* b1 = (const float*)d_in[10];
  const float* g2 = (const float*)d_in[11];
  const float* b2 = (const float*)d_in[12];
  const float* W1  = (const float*)d_in[13];
  const float* bl1 = (const float*)d_in[14];
  const float* W2  = (const float*)d_in[15];
  const float* bl2 = (const float*)d_in[16];
  (void)in_sizes; (void)n_in; (void)out_size; (void)ws_size;

  const long long* esrc = eidx;
  const long long* edst = eidx + (size_t)N_EDGES;

  char* ws = (char*)d_ws;
  float*     agg    = (float*)(ws + OFF_AGG);
  float*     h1     = (float*)(ws + OFF_H1);
  _Float16*  h2h    = (_Float16*)(ws + OFF_AGG);   // reuse agg region post-conv2
  _Float16*  W1h    = (_Float16*)(ws + OFF_W1H);
  _Float16*  W2h    = (_Float16*)(ws + OFF_W2H);
  _Float16*  mlp1h  = (_Float16*)(ws + OFF_MLP1);
  float*     stats  = (float*)(ws + OFF_STATS);    // 48 floats
  float*     stats1 = stats;
  float*     stats2 = stats + 24;

  const int TB = 256;

  // stats = 0
  zero_f32_kernel<<<1, 64, 0, stream>>>(stats, 48);

  // ---- conv1 ----
  zero_f32_kernel<<<16384, TB, 0, stream>>>(agg, (size_t)N_NODES * F_IN);
  scatter_add_kernel<F_IN><<<N_EDGES / TB, TB, 0, stream>>>(x, esrc, edst, agg);
  conv_linear_kernel<F_IN, F1><<<4096, TB, 0, stream>>>(agg, x, Wrel1, brel1, Wroot1, h1, stats1);
  bn_relu_kernel<false><<<8192, TB, 0, stream>>>(h1, stats1, g1, b1, h1, nullptr);

  // ---- conv2 ----
  zero_f32_kernel<<<16384, TB, 0, stream>>>(agg, (size_t)N_NODES * F1);
  scatter_add_kernel<F1><<<N_EDGES / TB, TB, 0, stream>>>(h1, esrc, edst, agg);
  conv_linear_kernel<F1, F2><<<4096, TB, 0, stream>>>(agg, h1, Wrel2, brel2, Wroot2, h1, stats2);
  bn_relu_kernel<true><<<8192, TB, 0, stream>>>(h1, stats2, g2, b2, nullptr, h2h);

  // ---- MLP head (f16 WMMA) ----
  cvt_f32_to_f16_kernel<<<2048, TB, 0, stream>>>(W1, W1h, (size_t)HID * K1);
  cvt_f32_to_f16_kernel<<<512,  TB, 0, stream>>>(W2, W2h, (size_t)NOUT * HID);

  // GEMM1: tiles = (4096/16) * (128/16) = 2048 waves, 8 waves/block -> 256 blocks
  gemm1_wmma_kernel<<<(BATCH / 16) * (HID / 16) / (TB / 32), TB, 0, stream>>>(h2h, W1h, bl1, mlp1h);
  // GEMM2: tiles = (4096/16) * (2048/16) = 32768 waves -> 4096 blocks
  gemm2_wmma_kernel<<<(BATCH / 16) * (NOUT / 16) / (TB / 32), TB, 0, stream>>>(mlp1h, W2h, bl2, (float*)d_out);
}
